// VQVAE_8005819039791
// MI455X (gfx1250) — matrix operand order repaired and checked by reading, not compiled, not run on hardware
//
#include <hip/hip_runtime.h>
#include <hip/hip_bf16.h>

typedef __attribute__((ext_vector_type(16))) __bf16 v16bf;
typedef __attribute__((ext_vector_type(8)))  float  v8f;

constexpr int Nn = 32768;
constexpr int Kk = 8192;
constexpr int Dd = 512;

// ---------------- Kernel 1: E -> bf16 hi/lo split + ||e||^2 (fp32) ----------------
__global__ __launch_bounds__(256) void vq_prep_kernel(const float* __restrict__ E,
                                                      __bf16* __restrict__ Ehi,
                                                      __bf16* __restrict__ Elo,
                                                      float* __restrict__ enorm) {
    const int row = blockIdx.x;
    const int tid = threadIdx.x;
    const float* er = E + (size_t)row * Dd;
    float s = 0.f;
    for (int i = tid; i < Dd; i += 256) {
        float e = er[i];
        __bf16 h = (__bf16)e;
        float hf = (float)h;
        __bf16 l = (__bf16)(e - hf);
        Ehi[(size_t)row * Dd + i] = h;
        Elo[(size_t)row * Dd + i] = l;
        s += e * e;
    }
    __shared__ float red[256];
    red[tid] = s;
    __syncthreads();
    for (int off = 128; off > 0; off >>= 1) {
        if (tid < off) red[tid] += red[tid + off];
        __syncthreads();
    }
    if (tid == 0) enorm[row] = red[0];
}

// ---------------- Kernel 2: WMMA bf16x3 GEMM + fused argmin ----------------
// 8 waves/WG, wave owns 16 rows of X (WG covers 128). K in chunks of 128 codes
// (8 accumulator tiles/wave). E chunks double-buffered in LDS via async-to-LDS DMA.
__global__ __launch_bounds__(256) void vq_argmin_kernel(const float* __restrict__ X,
                                                        const __bf16* __restrict__ Ehi,
                                                        const __bf16* __restrict__ Elo,
                                                        const float* __restrict__ enorm,
                                                        int* __restrict__ kidx) {
    constexpr int KC = 128;
    __shared__ uint4 smem[2][1024];  // per buffer: [0,512) hi plane, [512,1024) lo plane

    const int tid  = threadIdx.x;
    const int wave = tid >> 5;
    const int lane = tid & 31;
    const int half = lane >> 4;
    const int l16  = lane & 15;
    const int rowBase = blockIdx.x * 128 + wave * 16;
    const float* xrow = X + (size_t)(rowBase + l16) * Dd;

    const uint32_t ldsBase = (uint32_t)(size_t)(&smem[0][0]);
    const unsigned long long ehiB = (unsigned long long)(size_t)Ehi;
    const unsigned long long eloB = (unsigned long long)(size_t)Elo;

    float minv[8];
    int   mini[8];
#pragma unroll
    for (int v = 0; v < 8; ++v) { minv[v] = 3.4e38f; mini[v] = 0; }

    // async stage of E[kbase..+127][d..d+31] (hi+lo bf16 planes) into LDS buffer `buf`
    auto fillAsync = [&](int buf, int kbase, int d) {
        uint32_t lb = ldsBase + (uint32_t)buf * 16384u;
#pragma unroll
        for (int it = 0; it < 2; ++it) {
            int u  = tid + it * 256;          // 512 uint4 per plane
            int r  = u >> 2;                  // code row 0..127
            int c4 = u & 3;                   // 16B column within 64B row chunk
            uint32_t off = (uint32_t)((kbase + r) * Dd + d) * 2u + (uint32_t)c4 * 16u;
            uint32_t la  = lb + (uint32_t)u * 16u;
            asm volatile("global_load_async_to_lds_b128 %0, %1, %2"
                         :: "v"(la), "v"(off), "s"(ehiB) : "memory");
            asm volatile("global_load_async_to_lds_b128 %0, %1, %2"
                         :: "v"(la + 8192u), "v"(off), "s"(eloB) : "memory");
        }
    };

#pragma unroll 1
    for (int kbase = 0; kbase < Kk; kbase += KC) {
        v8f acc[8];
#pragma unroll
        for (int t = 0; t < 8; ++t) { v8f z = {}; acc[t] = z; }

        fillAsync(0, kbase, 0);  // prologue fill of buffer 0

#pragma unroll 1
        for (int dci = 0; dci < 16; ++dci) {
            const int d   = dci * 32;
            const int cur = dci & 1;

            asm volatile("s_wait_asynccnt 0x0" ::: "memory");  // my fills landed
            __syncthreads();  // everyone's fills landed; prior reads of other buffer done

            if (dci < 15) fillAsync(cur ^ 1, kbase, d + 32);  // overlap next DMA with WMMAs

            // A fragment: lane holds row (lane&15); K = half*16..+15 contiguous
            v16bf a_hi, a_lo;
            {
                const float* p = xrow + d + half * 16;
                float4 f0 = reinterpret_cast<const float4*>(p)[0];
                float4 f1 = reinterpret_cast<const float4*>(p)[1];
                const float* q = p + 8;
                float4 f2 = reinterpret_cast<const float4*>(q)[0];
                float4 f3 = reinterpret_cast<const float4*>(q)[1];
                float xv[16] = {f0.x, f0.y, f0.z, f0.w, f1.x, f1.y, f1.z, f1.w,
                                f2.x, f2.y, f2.z, f2.w, f3.x, f3.y, f3.z, f3.w};
#pragma unroll
                for (int i = 0; i < 16; ++i) {
                    __bf16 h = (__bf16)xv[i];
                    a_hi[i] = h;
                    a_lo[i] = (__bf16)(xv[i] - (float)h);
                }
            }

            // B fragments from LDS: lane holds column (lane&15), K = half*16..+15 contiguous
            const uint4* sm = &smem[cur][0];
#pragma unroll
            for (int t = 0; t < 8; ++t) {
                int c    = t * 16 + l16;
                int base = c * 4 + half * 2;
                union { uint4 u[2]; v16bf v; } bh, bl;
                bh.u[0] = sm[base];       bh.u[1] = sm[base + 1];
                bl.u[0] = sm[512 + base]; bl.u[1] = sm[512 + base + 1];
                acc[t] = __builtin_amdgcn_wmma_f32_16x16x32_bf16(false, a_hi, false, bh.v,
                                                                 (short)0, acc[t], false, false);
                acc[t] = __builtin_amdgcn_wmma_f32_16x16x32_bf16(false, a_hi, false, bl.v,
                                                                 (short)0, acc[t], false, false);
                acc[t] = __builtin_amdgcn_wmma_f32_16x16x32_bf16(false, a_lo, false, bh.v,
                                                                 (short)0, acc[t], false, false);
            }
        }

        // dist = ||e||^2 - 2*dot ; running elementwise argmin
#pragma unroll
        for (int t = 0; t < 8; ++t) {
            int cg = kbase + t * 16 + l16;
            float en = enorm[cg];
#pragma unroll
            for (int v = 0; v < 8; ++v) {
                float dist = fmaf(-2.f, acc[t][v], en);
                if (dist < minv[v]) { minv[v] = dist; mini[v] = cg; }
            }
        }
    }

    // cross-lane argmin within each 16-lane half (C layout: row M=v+8*half across lanes)
#pragma unroll
    for (int v = 0; v < 8; ++v) {
        float val = minv[v];
        int   idx = mini[v];
#pragma unroll
        for (int off = 1; off < 16; off <<= 1) {
            float ov = __shfl_xor(val, off, 16);
            int   oi = __shfl_xor(idx, off, 16);
            if (ov < val || (ov == val && oi < idx)) { val = ov; idx = oi; }
        }
        if (l16 == 0) kidx[rowBase + half * 8 + v] = idx;
    }
}

// ---------------- Kernel 3: gather + output assembly (x_q, z_e, z_q) ----------------
__global__ __launch_bounds__(128) void vq_output_kernel(const float* __restrict__ X,
                                                        const float* __restrict__ E,
                                                        const int* __restrict__ kidx,
                                                        float* __restrict__ out) {
    const int row = blockIdx.x;
    const int tid = threadIdx.x;  // 128 float4's per row
    const int k = kidx[row];
    const float4* e4 = reinterpret_cast<const float4*>(E + (size_t)k * Dd);
    const float4* x4 = reinterpret_cast<const float4*>(X + (size_t)row * Dd);
    float4* xq = reinterpret_cast<float4*>(out) + (size_t)row * 128;
    float4* ze = reinterpret_cast<float4*>(out + (size_t)Nn * Dd) + (size_t)row * 128;
    float4* zq = reinterpret_cast<float4*>(out + 2 * (size_t)Nn * Dd) + (size_t)row * 128;
    float4 ev = e4[tid];
    xq[tid] = ev;
    zq[tid] = ev;
    ze[tid] = x4[tid];
}

extern "C" void kernel_launch(void* const* d_in, const int* in_sizes, int n_in,
                              void* d_out, int out_size, void* d_ws, size_t ws_size,
                              hipStream_t stream) {
    const float* X = (const float*)d_in[0];
    const float* E = (const float*)d_in[1];
    float* out = (float*)d_out;

    char* ws = (char*)d_ws;
    __bf16* Ehi   = (__bf16*)(ws);
    __bf16* Elo   = (__bf16*)(ws + (size_t)Kk * Dd * 2);
    float*  enorm = (float*)(ws + (size_t)Kk * Dd * 4);
    int*    kidx  = (int*)(ws + (size_t)Kk * Dd * 4 + (size_t)Kk * 4);
    // ws usage: 8MB + 8MB + 32KB + 128KB ~= 16.2 MB

    vq_prep_kernel<<<Kk, 256, 0, stream>>>(E, Ehi, Elo, enorm);
    vq_argmin_kernel<<<Nn / 128, 256, 0, stream>>>(X, Ehi, Elo, enorm, kidx);
    vq_output_kernel<<<Nn, 128, 0, stream>>>(X, E, kidx, out);
}
